// CMMS_GCL_15848429322904
// MI455X (gfx1250) — matrix operand order repaired and checked by reading, not compiled, not run on hardware
//
#include <hip/hip_runtime.h>
#include <hip/hip_bf16.h>

typedef __attribute__((ext_vector_type(16))) __bf16 v16bf;
typedef __attribute__((ext_vector_type(8)))  __bf16 v8bf;
typedef __attribute__((ext_vector_type(8)))  float  v8f;
typedef __attribute__((ext_vector_type(4)))  unsigned int v4u;
typedef __attribute__((ext_vector_type(8)))  int v8i;
typedef __attribute__((ext_vector_type(4)))  int v4i;

// ---------------------------------------------------------------------------
// WMMA fragment loaders for PRE-PACKED bf16 operands (K padded to mult. of 32,
// zero-filled).  Rows are clamped (OOB rows feed OOB C rows which are never
// stored), so loads are unconditional 16-byte vector loads.
// A (16x32): lane L<16 -> row r0+L, K = kb..kb+7 , kb+16..kb+23 (kb=k0)
//            lane>=16  -> kb = k0+8
// B (32x16): lane L<16 -> W-row n0+L, K = kb..kb+7 , kb+16..kb+23 (kb=k0); lanes>=16 -> kb = k0+8
// C/D f32  : element (m0 + v + 8*(lane>>4), n0 + (lane&15)) in VGPR v
// ---------------------------------------------------------------------------
__device__ __forceinline__ v16bf load_a_frag(const __bf16* __restrict__ base,
                                             int r0, int Rm1, int k0, int ldp) {
  int lane = threadIdx.x & 31;
  int row = r0 + (lane & 15); if (row > Rm1) row = Rm1;
  int kb = k0 + ((lane >> 4) << 3);
  const __bf16* p = base + (long)row * ldp + kb;   // 16B aligned
  v8bf lo = *(const v8bf*)p;
  v8bf hi = *(const v8bf*)(p + 16);
  v16bf f;
#pragma unroll
  for (int i = 0; i < 8; ++i) { f[i] = lo[i]; f[i + 8] = hi[i]; }
  return f;
}

__device__ __forceinline__ v16bf load_b_frag(const __bf16* __restrict__ base,
                                             int r0, int Rm1, int k0, int ldp) {
  int lane = threadIdx.x & 31;
  int row = r0 + (lane & 15); if (row > Rm1) row = Rm1;
  int kb = k0 + ((lane >> 4) << 3);
  const __bf16* p = base + (long)row * ldp + kb;   // 16B aligned
  v8bf lo = *(const v8bf*)p;
  v8bf hi = *(const v8bf*)(p + 16);
  v16bf f;
#pragma unroll
  for (int i = 0; i < 8; ++i) { f[i] = lo[i]; f[i + 8] = hi[i]; }
  return f;
}

// B fragment from an LDS-staged 64x96 weight block (row-major, ld=96)
__device__ __forceinline__ v16bf load_b_frag_lds(const __bf16* sw, int r0, int k0) {
  int lane = threadIdx.x & 31;
  int row = r0 + (lane & 15);
  int kb = k0 + ((lane >> 4) << 3);
  const __bf16* p = sw + row * 96 + kb;            // 16B aligned -> ds_load_b128
  v8bf lo = *(const v8bf*)p;
  v8bf hi = *(const v8bf*)(p + 16);
  v16bf f;
#pragma unroll
  for (int i = 0; i < 8; ++i) { f[i] = lo[i]; f[i + 8] = hi[i]; }
  return f;
}

// Agent-scope coherent A loader (GRU hidden state: written by other waves,
// possibly on the other CU of the WGP, between steps). 8 dword atomic loads.
__device__ __forceinline__ v16bf load_a_frag_coh(const __bf16* __restrict__ base,
                                                 int r0, int Rm1, int k0, int ldp) {
  int lane = threadIdx.x & 31;
  int row = r0 + (lane & 15); if (row > Rm1) row = Rm1;
  int kb = k0 + ((lane >> 4) << 3);
  const unsigned int* p = (const unsigned int*)(base + (long)row * ldp + kb);
  unsigned int w0[4], w1[4];
#pragma unroll
  for (int i = 0; i < 4; ++i) {
    w0[i] = __hip_atomic_load(p + i,     __ATOMIC_RELAXED, __HIP_MEMORY_SCOPE_AGENT);
    w1[i] = __hip_atomic_load(p + 8 + i, __ATOMIC_RELAXED, __HIP_MEMORY_SCOPE_AGENT);
  }
  const __bf16* a0 = (const __bf16*)w0;
  const __bf16* a1 = (const __bf16*)w1;
  v16bf f;
#pragma unroll
  for (int i = 0; i < 8; ++i) { f[i] = a0[i]; f[i + 8] = a1[i]; }
  return f;
}

__device__ __forceinline__ v8f wmma_bf16(v16bf a, v16bf b, v8f c) {
  return __builtin_amdgcn_wmma_f32_16x16x32_bf16(false, a, false, b, (short)0, c, false, false);
}

// ---------------------------------------------------------------------------
// TDM: async 2-D tensor load of a bf16 tile (tile_rows x width, row-major,
// row stride = width elements) from global into LDS. D# per ISA ch.8.
// rows_valid = remaining rows in tensor (OOB rows are written as zeros).
// ---------------------------------------------------------------------------
__device__ __forceinline__ void tdm_load_2d_bf16(unsigned lds_off,
                                                 const __bf16* gsrc,
                                                 int rows_valid, int width,
                                                 int tile_rows) {
  unsigned long long ga = (unsigned long long)(const void*)gsrc;
  v4u g0;
  g0[0] = 1u;                                              // count=1, user D#
  g0[1] = lds_off;                                         // lds_addr (bytes)
  g0[2] = (unsigned)(ga & 0xFFFFFFFFu);                    // global_addr[31:0]
  g0[3] = (unsigned)((ga >> 32) & 0x1FFFFFFu) | (2u << 30); // addr[56:32]|type=2
  v8i g1;
  g1[0] = 0x10000;                                         // data_size=1 (2B)
  g1[1] = (int)((unsigned)width << 16);                    // tensor_dim0[15:0]
  g1[2] = (int)(((unsigned)rows_valid & 0xFFFFu) << 16);   // tensor_dim1[15:0]
  g1[3] = (int)((unsigned)width << 16);                    // tile_dim0
  g1[4] = tile_rows;                                       // tile_dim1 (tile_dim2=0)
  g1[5] = width;                                           // tensor_dim0_stride lo
  g1[6] = 0;
  g1[7] = 0;
  v4i gz = {0, 0, 0, 0};
#if defined(__clang_major__) && (__clang_major__ >= 23)
  v8i gz8 = {0, 0, 0, 0, 0, 0, 0, 0};
  __builtin_amdgcn_tensor_load_to_lds(g0, g1, gz, gz, gz8, 0);
#else
  __builtin_amdgcn_tensor_load_to_lds(g0, g1, gz, gz, 0);
#endif
}

// ---------------------------------------------------------------------------
// Generic GEMM on packed operands:
//   C(M,N) = act( A(M,Kp) @ W(N,Kp)^T + bias ),  Kp = padded K
// Optional outputs: Cf (f32, ld N) and/or Cb (bf16, ld KpOut, zero-padded).
// 4 waves/block, one 16x64 C strip per wave.  act: 0=none,1=relu,2=tanh
// ---------------------------------------------------------------------------
__global__ __launch_bounds__(128) void gemm_kernel(
    const __bf16* __restrict__ A, const __bf16* __restrict__ W,
    const float* __restrict__ bias,
    float* __restrict__ Cf, __bf16* __restrict__ Cb,
    int M, int N, int Kp, int KpOut, int act) {
  int wave = threadIdx.x >> 5;
  int lane = threadIdx.x & 31;
  int nblocks = (N + 63) >> 6;
  int nb = blockIdx.x * 4 + wave;
  if (nb >= nblocks) return;
  int m0 = blockIdx.y << 4;
  int n0 = nb << 6;
  v8f c0 = {}, c1 = {}, c2 = {}, c3 = {};
  for (int k0 = 0; k0 < Kp; k0 += 32) {
    v16bf a  = load_a_frag(A, m0, M - 1, k0, Kp);
    v16bf b0 = load_b_frag(W, n0,      N - 1, k0, Kp);
    v16bf b1 = load_b_frag(W, n0 + 16, N - 1, k0, Kp);
    v16bf b2 = load_b_frag(W, n0 + 32, N - 1, k0, Kp);
    v16bf b3 = load_b_frag(W, n0 + 48, N - 1, k0, Kp);
    c0 = wmma_bf16(a, b0, c0);
    c1 = wmma_bf16(a, b1, c1);
    c2 = wmma_bf16(a, b2, c2);
    c3 = wmma_bf16(a, b3, c3);
  }
  v8f cs[4] = {c0, c1, c2, c3};
  int mb = m0 + ((lane >> 4) << 3);
  int nc = lane & 15;
#pragma unroll
  for (int j = 0; j < 4; ++j) {
    int n = n0 + j * 16 + nc;
    float bv = (bias && n < N) ? bias[n] : 0.f;
#pragma unroll
    for (int v = 0; v < 8; ++v) {
      int m = mb + v;
      if (m >= M) continue;
      float val = 0.f;
      if (n < N) {
        val = cs[j][v] + bv;
        if (act == 1) val = fmaxf(val, 0.f);
        else if (act == 2) val = tanhf(val);
        if (Cf) Cf[(long)m * N + n] = val;
      }
      if (Cb && n < KpOut) Cb[(long)m * KpOut + n] = (__bf16)val;
    }
  }
}

// ---------------------------------------------------------------------------
// Pack f32 (R,K) [+ optional elementwise add of src2] -> bf16 (R,Kp), zero pad
// ---------------------------------------------------------------------------
__global__ void pack_bf16_kernel(const float* __restrict__ src,
                                 const float* __restrict__ src2,
                                 __bf16* __restrict__ dst,
                                 int R, int K, int Kp) {
  long i = (long)blockIdx.x * blockDim.x + threadIdx.x;
  if (i >= (long)R * Kp) return;
  int r = (int)(i / Kp);
  int k = (int)(i - (long)r * Kp);
  float v = 0.f;
  if (k < K) {
    long s = (long)r * K + k;
    v = src[s];
    if (src2) v += src2[s];
  }
  dst[i] = (__bf16)v;
}

// ---------------------------------------------------------------------------
// Persistent GRU direction: 1 workgroup, 32 waves, T=100 steps.
// gh = h @ wh^T + bh via WMMA into LDS (bf16); then gate update.
// h kept twice: f32 (recurrence math) + padded bf16 (matmul operand).
// ---------------------------------------------------------------------------
__global__ __launch_bounds__(1024) void gru_kernel(
    const float* __restrict__ gi, const __bf16* __restrict__ whb,
    const float* __restrict__ bh, float* __restrict__ hbuf,
    __bf16* __restrict__ hbf, float* __restrict__ outh, int rev) {
  __shared__ __bf16 sgh[100 * 304];     // 59.4 KB
  const int tid  = threadIdx.x;
  const int wave = tid >> 5, lane = tid & 31;
  for (int i = tid; i < 100 * 100; i += 1024) hbuf[i] = 0.f;
  for (int i = tid; i < (100 * 128) / 2; i += 1024) ((unsigned int*)hbf)[i] = 0u;
  __threadfence();
  __syncthreads();
  for (int s = 0; s < 100; ++s) {
    const int trow = rev ? (99 - s) : s;
    // Phase A: gh tiles (7 x 5 strips of 16x64)
    for (int task = wave; task < 35; task += 32) {
      int mt = task / 5, nbk = task % 5;
      int m0 = mt * 16, n0 = nbk * 64;
      v8f c0 = {}, c1 = {}, c2 = {}, c3 = {};
      for (int k0 = 0; k0 < 128; k0 += 32) {
        v16bf a  = load_a_frag_coh(hbf, m0, 99, k0, 128);
        v16bf b0 = load_b_frag(whb, n0,      299, k0, 128);
        v16bf b1 = load_b_frag(whb, n0 + 16, 299, k0, 128);
        v16bf b2 = load_b_frag(whb, n0 + 32, 299, k0, 128);
        v16bf b3 = load_b_frag(whb, n0 + 48, 299, k0, 128);
        c0 = wmma_bf16(a, b0, c0);
        c1 = wmma_bf16(a, b1, c1);
        c2 = wmma_bf16(a, b2, c2);
        c3 = wmma_bf16(a, b3, c3);
      }
      v8f cs[4] = {c0, c1, c2, c3};
      int mb = m0 + ((lane >> 4) << 3);
      int nc = lane & 15;
#pragma unroll
      for (int j = 0; j < 4; ++j) {
        int n = n0 + j * 16 + nc;
        if (n >= 304) continue;
        float bv = (n < 300) ? bh[n] : 0.f;
#pragma unroll
        for (int v = 0; v < 8; ++v) {
          int m = mb + v;
          if (m < 100) sgh[m * 304 + n] = (__bf16)(cs[j][v] + bv);
        }
      }
    }
    __syncthreads();
    // Phase B: gate update
    for (int i = tid; i < 100 * 100; i += 1024) {
      int b = i / 100, j = i - b * 100;
      const float* g = gi + ((long)trow * 100 + b) * 300;
      float hr = (float)sgh[b * 304 + j];
      float hz = (float)sgh[b * 304 + 100 + j];
      float hn = (float)sgh[b * 304 + 200 + j];
      float r = 1.f / (1.f + __expf(-(g[j]       + hr)));
      float z = 1.f / (1.f + __expf(-(g[100 + j] + hz)));
      float n = tanhf(g[200 + j] + r * hn);
      float hnew = (1.f - z) * n + z * hbuf[i];
      hbuf[i] = hnew;
      hbf[b * 128 + j] = (__bf16)hnew;
      outh[((long)trow * 100 + b) * 100 + j] = hnew;
    }
    __threadfence();
    __syncthreads();
  }
}

// ---------------------------------------------------------------------------
// Fused GIN conv2 GEMM + max/mean pooling. One workgroup per graph
// (contiguous node segments). Loops over 64-column weight blocks; each block
// (64x96 bf16, 12KB) is staged into LDS once via the Tensor Data Mover, then
// all node tiles of the graph are processed against it (B from LDS).
// Values are post-ReLU >= 0 so float max == int max on bit patterns.
// ---------------------------------------------------------------------------
__global__ __launch_bounds__(256) void gin2_pool_kernel(
    const __bf16* __restrict__ h1a, const __bf16* __restrict__ w2b,
    const float* __restrict__ b2, float* __restrict__ pooled, int N, int G) {
  __shared__ __bf16 sw[64 * 96];        // 12 KB staged weight block
  __shared__ float smax[840];
  __shared__ float ssum[840];
  int g = blockIdx.x;
  int start = (int)(((long)g * N + G - 1) / G);
  int end   = (int)(((long)(g + 1) * N + G - 1) / G);
  int cnt = end - start;
  int tid = threadIdx.x, wave = tid >> 5, lane = tid & 31;
  for (int i = tid; i < 840; i += 256) { smax[i] = 0.f; ssum[i] = 0.f; }
  __syncthreads();
  int mtiles = (cnt + 15) >> 4;
  for (int nbk = 0; nbk < 14; ++nbk) {
    int n0 = nbk * 64;
    if (wave == 0) {                    // one TDM descriptor per weight block
      tdm_load_2d_bf16((unsigned)(size_t)&sw[0], w2b + (long)n0 * 96,
                       840 - n0, 96, 64);
      __builtin_amdgcn_s_wait_tensorcnt(0);
    }
    __syncthreads();                    // LDS tile visible to all waves
    for (int task = wave; task < mtiles; task += 8) {
      int m0 = start + task * 16;
      v8f c0 = {}, c1 = {}, c2 = {}, c3 = {};
      for (int k0 = 0; k0 < 96; k0 += 32) {
        v16bf a  = load_a_frag(h1a, m0, end - 1, k0, 96);
        v16bf b0 = load_b_frag_lds(sw, 0,  k0);
        v16bf b1 = load_b_frag_lds(sw, 16, k0);
        v16bf b2 = load_b_frag_lds(sw, 32, k0);
        v16bf b3 = load_b_frag_lds(sw, 48, k0);
        c0 = wmma_bf16(a, b0, c0);
        c1 = wmma_bf16(a, b1, c1);
        c2 = wmma_bf16(a, b2, c2);
        c3 = wmma_bf16(a, b3, c3);
      }
      v8f cs[4] = {c0, c1, c2, c3};
      int mb = m0 + ((lane >> 4) << 3);
      int nc = lane & 15;
#pragma unroll
      for (int j = 0; j < 4; ++j) {
        int n = n0 + j * 16 + nc;
        if (n >= 840) continue;
        float bv = b2[n];
#pragma unroll
        for (int v = 0; v < 8; ++v) {
          int m = mb + v;
          if (m < end) {
            float val = fmaxf(cs[j][v] + bv, 0.f);
            atomicMax((int*)&smax[n], __float_as_int(val));
            atomicAdd(&ssum[n], val);
          }
        }
      }
    }
    __syncthreads();                    // done with sw before restaging
  }
  for (int i = tid; i < 840; i += 256) {
    pooled[(long)g * 1680 + i]       = smax[i];
    pooled[(long)g * 1680 + 840 + i] = ssum[i] / (float)cnt;
  }
}

// ---------------------------------------------------------------------------
// Elementwise / reduction helper kernels
// ---------------------------------------------------------------------------
__global__ void zero_kernel(float* __restrict__ p, long n) {
  long i = (long)blockIdx.x * blockDim.x + threadIdx.x;
  if (i < n) p[i] = 0.f;
}

__global__ void scatter_add_kernel(const float* __restrict__ h,
                                   const int* __restrict__ ei,
                                   float* __restrict__ agg, int E, int F) {
  int e = blockIdx.x * blockDim.x + threadIdx.x;
  if (e >= E) return;
  int src = ei[e];
  int dst = ei[E + e];
  const float* ps = h + (long)src * F;
  float* pd = agg + (long)dst * F;
  __builtin_prefetch(ps, 0, 1);
  for (int f = 0; f < F; ++f) atomicAdd(&pd[f], ps[f]);
}

__global__ void mask_kernel(const float* __restrict__ x,
                            const float* __restrict__ mask,
                            float* __restrict__ out, long total, int F) {
  long i = (long)blockIdx.x * blockDim.x + threadIdx.x;
  if (i < total) out[i] = x[i] * mask[i / F];
}

// s = relu(concat(hf,hb)) -> f32 (10000,200) and packed bf16 (10000,224)
__global__ void build_s_kernel(const float* __restrict__ hf,
                               const float* __restrict__ hb,
                               float* __restrict__ s, __bf16* __restrict__ sbf) {
  long i = (long)blockIdx.x * blockDim.x + threadIdx.x;
  if (i >= 10000L * 224) return;
  long tb = i / 224; int c = (int)(i - tb * 224);
  float v = 0.f;
  if (c < 200) {
    v = (c < 100) ? hf[tb * 100 + c] : hb[tb * 100 + (c - 100)];
    v = fmaxf(v, 0.f);
    s[tb * 200 + c] = v;
  }
  sbf[i] = (__bf16)v;
}

__global__ void softmax_b_kernel(float* __restrict__ att) {
  int idx = blockIdx.x * blockDim.x + threadIdx.x;   // (t,c)
  if (idx >= 100 * 256) return;
  int t = idx / 256, c = idx - (idx / 256) * 256;
  float* base = att + (long)t * 100 * 256 + c;
  float mx = -3.4e38f;
  for (int b = 0; b < 100; ++b) mx = fmaxf(mx, base[b * 256]);
  float ssum = 0.f;
  for (int b = 0; b < 100; ++b) ssum += __expf(base[b * 256] - mx);
  float inv = 1.f / ssum;
  for (int b = 0; b < 100; ++b) base[b * 256] = __expf(base[b * 256] - mx) * inv;
}

__global__ void att_rowsum_kernel(const float* __restrict__ att, float* __restrict__ wsum) {
  int i = blockIdx.x * blockDim.x + threadIdx.x;     // t*100+b
  if (i >= 10000) return;
  const float* p = att + (long)i * 256;
  float s = 0.f;
  for (int c = 0; c < 256; ++c) s += p[c];
  wsum[i] = s;
}

__global__ void att_apply_kernel(const float* __restrict__ wsum,
                                 const float* __restrict__ s,
                                 float* __restrict__ smi0) {
  int i = blockIdx.x * blockDim.x + threadIdx.x;     // t*200+d
  if (i >= 100 * 200) return;
  int t = i / 200, d = i - t * 200;
  float acc = 0.f;
  for (int b = 0; b < 100; ++b)
    acc += wsum[t * 100 + b] * s[((long)t * 100 + b) * 200 + d];
  smi0[i] = acc * 0.1f;
}

// cat = concat(x_g, smi) packed straight to bf16 (768 is a multiple of 32)
__global__ void build_cat_kernel(const float* __restrict__ xg,
                                 const float* __restrict__ smi,
                                 __bf16* __restrict__ cat) {
  int i = blockIdx.x * blockDim.x + threadIdx.x;
  if (i >= 100 * 768) return;
  int r = i / 768, c = i - r * 768;
  float v = (c < 512) ? xg[r * 512 + c] : smi[r * 256 + (c - 512)];
  cat[i] = (__bf16)v;
}

__global__ void final_dot_kernel(const float* __restrict__ f1,
                                 const float* __restrict__ w,
                                 const float* __restrict__ b,
                                 float* __restrict__ z) {
  int t = blockIdx.x * blockDim.x + threadIdx.x;
  if (t >= 100) return;
  float acc = b[0];
  for (int c = 0; c < 256; ++c) acc += f1[t * 256 + c] * w[c];
  z[t] = acc;
}

// ---------------------------------------------------------------------------
extern "C" void kernel_launch(void* const* d_in, const int* in_sizes, int n_in,
                              void* d_out, int out_size, void* d_ws, size_t ws_size,
                              hipStream_t stream) {
  (void)in_sizes; (void)n_in; (void)out_size; (void)ws_size;
  const int N = 150000, E = 1200000, G = 100, F = 84;

  const float* x         = (const float*)d_in[0];
  const float* smi_em    = (const float*)d_in[1];
  const float* node_mask = (const float*)d_in[2];
  const int*   edge_idx  = (const int*)d_in[3];
  // d_in[4] (batch) not needed: segments contiguous, computed analytically
  const float* gru_wi_f = (const float*)d_in[5];
  const float* gru_wh_f = (const float*)d_in[6];
  const float* gru_bi_f = (const float*)d_in[7];
  const float* gru_bh_f = (const float*)d_in[8];
  const float* gru_wi_b = (const float*)d_in[9];
  const float* gru_wh_b = (const float*)d_in[10];
  const float* gru_bi_b = (const float*)d_in[11];
  const float* gru_bh_b = (const float*)d_in[12];
  const float* fc_w1 = (const float*)d_in[13];
  const float* fc_b1 = (const float*)d_in[14];
  const float* fc_w2 = (const float*)d_in[15];
  const float* fc_b2 = (const float*)d_in[16];
  const float* lin_w1 = (const float*)d_in[17];
  const float* lin_b1 = (const float*)d_in[18];
  const float* lin_w2 = (const float*)d_in[19];
  const float* lin_b2 = (const float*)d_in[20];
  const float* conv1_w = (const float*)d_in[21];
  const float* conv1_b = (const float*)d_in[22];
  const float* conv2_w = (const float*)d_in[23];
  const float* conv2_b = (const float*)d_in[24];
  const float* fcg_w1 = (const float*)d_in[25];
  const float* fcg_b1 = (const float*)d_in[26];
  const float* fcg_w2 = (const float*)d_in[27];
  const float* fcg_b2 = (const float*)d_in[28];
  const float* ff_w1 = (const float*)d_in[29];
  const float* ff_b1 = (const float*)d_in[30];
  const float* ff_w2 = (const float*)d_in[31];
  const float* ff_b2 = (const float*)d_in[32];

  float* outp = (float*)d_out;       // [0,100)=z, [100,51300)=x_g, [51300,102500)=y_g
  float* xg_out = outp + 100;
  float* yg_out = outp + 100 + 100 * 512;

  // ---- workspace: f32 region then bf16 region ----
  float* ws = (float*)d_ws;
  size_t o = 0;
  auto allocf = [&](size_t n) { float* p = ws + o; o += n; return p; };
  float* gi_f  = allocf(10000L * 300);
  float* gi_b  = allocf(10000L * 300);
  float* hbuf  = allocf(10000);
  float* hfw   = allocf(1000000);
  float* hbw   = allocf(1000000);
  float* sbuf  = allocf(2000000);
  float* att   = allocf(10000L * 256);
  float* wsum  = allocf(10000);
  float* smi0  = allocf(100L * 200);
  float* smi   = allocf(100L * 256);
  float* xmask = allocf((long)N * F);
  float* agg   = allocf((long)N * F);
  float* h1    = allocf((long)N * F);
  float* pooled= allocf(100L * 1680);
  float* f1    = allocf(100L * 256);
  o = (o + 7) & ~(size_t)7;                       // 16B-align bf16 region
  __bf16* wb = (__bf16*)(ws + o);
  size_t ob = 0;
  auto allocb = [&](size_t n) { __bf16* p = wb + ob; ob += (n + 7) & ~(size_t)7; return p; };
  __bf16* smi_em_bf = allocb(10000L * 128);
  __bf16* wi_f_bf   = allocb(300L * 128);
  __bf16* wh_f_bf   = allocb(300L * 128);
  __bf16* wi_b_bf   = allocb(300L * 128);
  __bf16* wh_b_bf   = allocb(300L * 128);
  __bf16* hbf       = allocb(100L * 128);
  __bf16* sbuf_bf   = allocb(10000L * 224);
  __bf16* fc_w1_bf  = allocb(512L * 224);
  __bf16* u_bf      = allocb(10000L * 512);
  __bf16* fc_w2_bf  = allocb(256L * 512);
  __bf16* smi0_bf   = allocb(100L * 224);
  __bf16* lin_w1_bf = allocb(512L * 224);
  __bf16* smi1_bf   = allocb(100L * 512);
  __bf16* lin_w2_bf = allocb(256L * 512);
  __bf16* conv1_w_bf= allocb(84L * 96);
  __bf16* conv2_w_bf= allocb(840L * 96);
  __bf16* xa_bf     = allocb((long)N * 96);
  __bf16* h1a_bf    = allocb((long)N * 96);
  __bf16* pooled_bf = allocb(100L * 1696);
  __bf16* fcg_w1_bf = allocb(1024L * 1696);
  __bf16* fcg1_bf   = allocb(100L * 1024);
  __bf16* fcg_w2_bf = allocb(512L * 1024);
  __bf16* cat_bf    = allocb(100L * 768);
  __bf16* ff_w1_bf  = allocb(256L * 768);

  auto pack = [&](const float* src, const float* src2, __bf16* dst, int R, int K, int Kp) {
    long tot = (long)R * Kp;
    pack_bf16_kernel<<<(int)((tot + 255) / 256), 256, 0, stream>>>(src, src2, dst, R, K, Kp);
  };
  auto gemm = [&](const __bf16* A, const __bf16* W, const float* bias,
                  float* Cf, __bf16* Cb, int M, int Nn, int Kp, int KpOut, int act) {
    int nblocks = (Nn + 63) / 64;
    dim3 grid((nblocks + 3) / 4, (M + 15) / 16);
    gemm_kernel<<<grid, 128, 0, stream>>>(A, W, bias, Cf, Cb, M, Nn, Kp, KpOut, act);
  };

  // ---- pack weights ----
  pack(smi_em, nullptr, smi_em_bf, 10000, 100, 128);
  pack(gru_wi_f, nullptr, wi_f_bf, 300, 100, 128);
  pack(gru_wh_f, nullptr, wh_f_bf, 300, 100, 128);
  pack(gru_wi_b, nullptr, wi_b_bf, 300, 100, 128);
  pack(gru_wh_b, nullptr, wh_b_bf, 300, 100, 128);
  pack(fc_w1, nullptr, fc_w1_bf, 512, 200, 224);
  pack(fc_w2, nullptr, fc_w2_bf, 256, 512, 512);
  pack(lin_w1, nullptr, lin_w1_bf, 512, 200, 224);
  pack(lin_w2, nullptr, lin_w2_bf, 256, 512, 512);
  pack(conv1_w, nullptr, conv1_w_bf, 84, 84, 96);
  pack(conv2_w, nullptr, conv2_w_bf, 840, 84, 96);
  pack(fcg_w1, nullptr, fcg_w1_bf, 1024, 1680, 1696);
  pack(fcg_w2, nullptr, fcg_w2_bf, 512, 1024, 1024);
  pack(ff_w1, nullptr, ff_w1_bf, 256, 768, 768);

  // ---- SMILES branch ----
  gemm(smi_em_bf, wi_f_bf, gru_bi_f, gi_f, nullptr, 10000, 300, 128, 0, 0);
  gemm(smi_em_bf, wi_b_bf, gru_bi_b, gi_b, nullptr, 10000, 300, 128, 0, 0);
  gru_kernel<<<1, 1024, 0, stream>>>(gi_f, wh_f_bf, gru_bh_f, hbuf, hbf, hfw, 0);
  gru_kernel<<<1, 1024, 0, stream>>>(gi_b, wh_b_bf, gru_bh_b, hbuf, hbf, hbw, 1);
  build_s_kernel<<<(int)((10000L * 224 + 255) / 256), 256, 0, stream>>>(hfw, hbw, sbuf, sbuf_bf);
  gemm(sbuf_bf, fc_w1_bf, fc_b1, nullptr, u_bf, 10000, 512, 224, 512, 1);
  gemm(u_bf, fc_w2_bf, fc_b2, att, nullptr, 10000, 256, 512, 0, 2);
  softmax_b_kernel<<<(25600 + 255) / 256, 256, 0, stream>>>(att);
  att_rowsum_kernel<<<(10000 + 255) / 256, 256, 0, stream>>>(att, wsum);
  att_apply_kernel<<<(20000 + 255) / 256, 256, 0, stream>>>(wsum, sbuf, smi0);
  pack(smi0, nullptr, smi0_bf, 100, 200, 224);
  gemm(smi0_bf, lin_w1_bf, lin_b1, nullptr, smi1_bf, 100, 512, 224, 512, 0);
  gemm(smi1_bf, lin_w2_bf, lin_b2, smi, nullptr, 100, 256, 512, 0, 0);

  // ---- graph branches ----
  long totNF = (long)N * F;
  mask_kernel<<<(int)((totNF + 255) / 256), 256, 0, stream>>>(x, node_mask, xmask, totNF, F);

  const float* Xb[2] = {x, xmask};
  float* Ob[2] = {xg_out, yg_out};
  for (int br = 0; br < 2; ++br) {
    zero_kernel<<<(int)((totNF + 255) / 256), 256, 0, stream>>>(agg, totNF);
    scatter_add_kernel<<<(E + 255) / 256, 256, 0, stream>>>(Xb[br], edge_idx, agg, E, F);
    pack(Xb[br], agg, xa_bf, N, F, 96);
    gemm(xa_bf, conv1_w_bf, conv1_b, h1, nullptr, N, F, 96, 0, 1);
    zero_kernel<<<(int)((totNF + 255) / 256), 256, 0, stream>>>(agg, totNF);
    scatter_add_kernel<<<(E + 255) / 256, 256, 0, stream>>>(h1, edge_idx, agg, E, F);
    pack(h1, agg, h1a_bf, N, F, 96);
    gin2_pool_kernel<<<G, 256, 0, stream>>>(h1a_bf, conv2_w_bf, conv2_b, pooled, N, G);
    pack(pooled, nullptr, pooled_bf, 100, 1680, 1696);
    gemm(pooled_bf, fcg_w1_bf, fcg_b1, nullptr, fcg1_bf, 100, 1024, 1696, 1024, 1);
    gemm(fcg1_bf, fcg_w2_bf, fcg_b2, Ob[br], nullptr, 100, 512, 1024, 0, 0);
  }

  // ---- head ----
  build_cat_kernel<<<(76800 + 255) / 256, 256, 0, stream>>>(xg_out, smi, cat_bf);
  gemm(cat_bf, ff_w1_bf, ff_b1, f1, nullptr, 100, 256, 768, 0, 1);
  final_dot_kernel<<<1, 128, 0, stream>>>(f1, ff_w2, ff_b2, outp);
}
